// TransformerBlock_66949950210198
// MI455X (gfx1250) — matrix operand order repaired and checked by reading, not compiled, not run on hardware
//
#include <hip/hip_runtime.h>
#include <hip/hip_bf16.h>
#include <math.h>
#include <stdint.h>

// ---------------------------------------------------------------------------
// Transformer block for MI455X (gfx1250, wave32, WMMA + Tensor Data Mover).
//
// Precision strategy: all GEMMs in bf16 (v_wmma_f32_16x16x32_bf16, fp32
// accumulate). fp32 operands are pre-converted to bf16 once (~300MB traffic,
// ~13us @ 23.3TB/s) which halves GEMM-side HBM reads; the GEMM inner loop
// then uses TDM (tensor_load_to_lds, TENSORcnt) to DMA 128x32 bf16 tiles
// into LDS, double-buffered so DMA of tile k+1 overlaps WMMA on tile k.
// ---------------------------------------------------------------------------

#define EPSF 1.1920929e-07f

typedef __attribute__((ext_vector_type(16))) __bf16        v16bf;
typedef __attribute__((ext_vector_type(8)))  float         v8f;
typedef __attribute__((ext_vector_type(4)))  unsigned int  v4u;
typedef __attribute__((ext_vector_type(8)))  int           v8i;
typedef __attribute__((ext_vector_type(4)))  int           v4i;

union FragU {
    v16bf v;
    unsigned int u[8];
};

// fp32 -> bf16 round-to-nearest-even
__device__ __forceinline__ unsigned short bf16_rne(float a) {
    unsigned int u = __float_as_uint(a);
    u += 0x7FFFu + ((u >> 16) & 1u);
    return (unsigned short)(u >> 16);
}
__device__ __forceinline__ unsigned int pack2_bf16(float a, float b) {
    unsigned int ua = __float_as_uint(a), ub = __float_as_uint(b);
    ua += 0x7FFFu + ((ua >> 16) & 1u);
    ub += 0x7FFFu + ((ub >> 16) & 1u);
    return (ua >> 16) | (ub & 0xFFFF0000u);
}

__device__ __forceinline__ float warpReduceSum(float v) {
#pragma unroll
    for (int o = 16; o > 0; o >>= 1) v += __shfl_xor(v, o, 32);
    return v;
}
__device__ __forceinline__ float warpReduceMax(float v) {
#pragma unroll
    for (int o = 16; o > 0; o >>= 1) v = fmaxf(v, __shfl_xor(v, o, 32));
    return v;
}
__device__ __forceinline__ float blockReduceSum(float v, float* sm) {
    int lane = threadIdx.x & 31, w = threadIdx.x >> 5;
    __syncthreads();
    v = warpReduceSum(v);
    if (lane == 0) sm[w] = v;
    __syncthreads();
    if (w == 0) {
        float r = (lane < 8) ? sm[lane] : 0.0f;
        r = warpReduceSum(r);
        if (lane == 0) sm[0] = r;
    }
    __syncthreads();
    return sm[0];
}
__device__ __forceinline__ float blockReduceMax(float v, float* sm) {
    int lane = threadIdx.x & 31, w = threadIdx.x >> 5;
    __syncthreads();
    v = warpReduceMax(v);
    if (lane == 0) sm[w] = v;
    __syncthreads();
    if (w == 0) {
        float r = (lane < 8) ? sm[lane] : -3.402823466e38f;
        r = warpReduceMax(r);
        if (lane == 0) sm[0] = r;
    }
    __syncthreads();
    return sm[0];
}

// ---------------------------------------------------------------------------
// Tensor Data Mover: DMA a [rows=128 x k=32] bf16 tile (row stride = ld
// elements) from global memory into LDS at lds_off, with D# padding of
// 4 DWORDs every 16 DWORDs -> LDS row stride 40 bf16 elements (80B), which
// keeps the 16-row WMMA fragment reads bank-conflict-free.
// D# layout per CDNA5 ISA 8.3-8.5. Tracked by TENSORcnt.
// ---------------------------------------------------------------------------
__device__ __forceinline__ void tdm_load_tile(unsigned lds_off,
                                              unsigned long long gaddr,
                                              unsigned long long ld_elems)
{
    v4u g0;
    g0[0] = 1u;                                   // count=1, user descriptor
    g0[1] = lds_off;                              // lds_addr (bytes)
    g0[2] = (unsigned)(gaddr & 0xFFFFFFFFull);    // global_addr[31:0]
    g0[3] = ((unsigned)(gaddr >> 32) & 0x01FFFFFFu) | (2u << 30); // [56:32]|type=2

    v8i g1;
    // data_size=2B(1<<16) | pad_enable(1<<20) | pad_interval=16DW(3<<22)
    // | pad_amount=4DW(3<<25)
    g1[0] = (int)((1u << 16) | (1u << 20) | (3u << 22) | (3u << 25));
    const unsigned td0 = 0x40000000u, td1 = 0x40000000u;  // huge: no OOB clip
    g1[1] = (int)((td0 & 0xFFFFu) << 16);                 // tensor_dim0[15:0]
    g1[2] = (int)((td0 >> 16) | ((td1 & 0xFFFFu) << 16)); // td0 hi | td1 lo
    g1[3] = (int)((td1 >> 16) | (32u << 16));             // td1 hi | tile_dim0=32
    g1[4] = (int)(128u);                                  // tile_dim1=128 rows
    g1[5] = (int)(unsigned)(ld_elems & 0xFFFFFFFFull);    // dim0_stride lo
    g1[6] = (int)(unsigned)((ld_elems >> 32) & 0xFFFFull);// dim0_stride hi
    g1[7] = 0;

    v4i g2 = {0, 0, 0, 0};
    v4i g3 = {0, 0, 0, 0};
#if __has_include(<hip/amd_detail/amd_gfx1250_TDM.h>)
    v8i g4 = {0, 0, 0, 0, 0, 0, 0, 0};
    __builtin_amdgcn_tensor_load_to_lds(g0, g1, g2, g3, g4, 0);
#else
    __builtin_amdgcn_tensor_load_to_lds(g0, g1, g2, g3, 0);
#endif
}

__device__ __forceinline__ unsigned lds_byte_off(const void* p) {
    // generic LDS pointer: addr[31:0] is the LDS byte offset (ISA 10.2)
    return (unsigned)(unsigned long long)(uintptr_t)p;
}

// ---------------------------------------------------------------------------
// NT GEMM on bf16: C[M,N] = alpha * A[M,K] * B[N,K]^T, epilogue by template.
// 256 threads = 8 waves; block tile 128x128; wave tile 64x32 (4x2 WMMA).
// TDM double-buffered LDS staging; 8 v_wmma_f32_16x16x32_bf16 per K-step.
// ---------------------------------------------------------------------------
#define LDSS 40  // LDS row stride in bf16 elements (32 data + 4DW TDM pad)

enum { OM_BF16 = 0, OM_BF16_T = 1, OM_F32 = 2, OM_F32_BIAS = 3, OM_F32_BIAS_RES = 4 };

template <int OM>
__global__ __launch_bounds__(256) void gemm_tdm_wmma(
    const unsigned short* __restrict__ A, long lda,   // bf16 [M,K]
    const unsigned short* __restrict__ Bm, long ldb,  // bf16 [N,K]
    void* __restrict__ out, long ldo,
    int K, float alpha,
    const float* __restrict__ bias,
    const float* __restrict__ residual, long ldr)
{
    __shared__ unsigned short As[2][128 * LDSS];
    __shared__ unsigned short Bs[2][128 * LDSS];

    const int tid   = threadIdx.x;
    const int lane  = tid & 31;
    const int wave  = tid >> 5;
    const int waveM = wave >> 2;   // 0..1
    const int waveN = wave & 3;    // 0..3
    const int lh    = lane >> 4;   // K-phase half
    const int lr    = lane & 15;   // row/col within 16

    const long bm = (long)blockIdx.y * 128;
    const long bn = (long)blockIdx.x * 128;

    const unsigned long long aBase =
        (unsigned long long)(uintptr_t)A + (unsigned long long)bm * lda * 2ull;
    const unsigned long long bBase =
        (unsigned long long)(uintptr_t)Bm + (unsigned long long)bn * ldb * 2ull;

    v8f acc[4][2];
#pragma unroll
    for (int i = 0; i < 4; ++i)
#pragma unroll
        for (int j = 0; j < 2; ++j) acc[i][j] = {};

    // prime buffer 0 (wave 0 issues the DMA; TDM ignores EXEC)
    if (tid < 32) {
        tdm_load_tile(lds_byte_off(&As[0][0]), aBase, (unsigned long long)lda);
        tdm_load_tile(lds_byte_off(&Bs[0][0]), bBase, (unsigned long long)ldb);
    }

    int buf = 0;
    for (int k0 = 0; k0 < K; k0 += 32) {
        if (tid < 32) {
            if (k0 + 32 < K) {
                // kick off next tile into the other buffer, then wait for the
                // current pair only (in-order completion => <=2 outstanding)
                tdm_load_tile(lds_byte_off(&As[buf ^ 1][0]),
                              aBase + (unsigned long long)(k0 + 32) * 2ull,
                              (unsigned long long)lda);
                tdm_load_tile(lds_byte_off(&Bs[buf ^ 1][0]),
                              bBase + (unsigned long long)(k0 + 32) * 2ull,
                              (unsigned long long)ldb);
                __builtin_amdgcn_s_wait_tensorcnt(2);
            } else {
                __builtin_amdgcn_s_wait_tensorcnt(0);
            }
        }
        __syncthreads();

        // A fragments: 16x32 bf16 (lane half h holds K[h*16..h*16+15],
        // VGPR0-3 K[h*16..h*16+7], VGPR4-7 K[h*16+8..h*16+15])
        FragU af[4];
#pragma unroll
        for (int i = 0; i < 4; ++i) {
            const unsigned int* p = (const unsigned int*)(
                &As[buf][(waveM * 64 + i * 16 + lr) * LDSS]);
            const int o0 = lh * 8, o1 = lh * 8 + 4;
            af[i].u[0] = p[o0 + 0]; af[i].u[1] = p[o0 + 1];
            af[i].u[2] = p[o0 + 2]; af[i].u[3] = p[o0 + 3];
            af[i].u[4] = p[o1 + 0]; af[i].u[5] = p[o1 + 1];
            af[i].u[6] = p[o1 + 2]; af[i].u[7] = p[o1 + 3];
        }
        // B fragments: 32x16 bf16 (lanes 0-15 K=0..15, lanes 16-31 K=16..31)
        FragU bfr[2];
#pragma unroll
        for (int j = 0; j < 2; ++j) {
            const unsigned int* p = (const unsigned int*)(
                &Bs[buf][(waveN * 32 + j * 16 + lr) * LDSS]);
            const int o = lh * 8;
#pragma unroll
            for (int t = 0; t < 8; ++t) bfr[j].u[t] = p[o + t];
        }

#pragma unroll
        for (int i = 0; i < 4; ++i)
#pragma unroll
            for (int j = 0; j < 2; ++j)
                acc[i][j] = __builtin_amdgcn_wmma_f32_16x16x32_bf16(
                    false, af[i].v, false, bfr[j].v,
                    (short)0, acc[i][j], false, false);

        __syncthreads();
        buf ^= 1;
    }

    // epilogue (branch-free per template) — C layout: VGPR v -> row
    // v + (lane/16)*8, col lane%16
#pragma unroll
    for (int i = 0; i < 4; ++i) {
#pragma unroll
        for (int j = 0; j < 2; ++j) {
            const long n = bn + waveN * 32 + j * 16 + lr;
            float bv = 0.0f;
            if (OM == OM_F32_BIAS || OM == OM_F32_BIAS_RES) bv = bias[n];
#pragma unroll
            for (int v = 0; v < 8; ++v) {
                const long m = bm + waveM * 64 + i * 16 + lh * 8 + v;
                float cv = acc[i][j][v] * alpha;
                if (OM == OM_BF16) {
                    ((unsigned short*)out)[m * ldo + n] = bf16_rne(cv);
                } else if (OM == OM_BF16_T) {
                    ((unsigned short*)out)[n * ldo + m] = bf16_rne(cv);
                } else if (OM == OM_F32) {
                    ((float*)out)[m * ldo + n] = cv;
                } else if (OM == OM_F32_BIAS) {
                    ((float*)out)[m * ldo + n] = cv + bv;
                } else {  // OM_F32_BIAS_RES
                    ((float*)out)[m * ldo + n] = cv + bv + residual[m * ldr + n];
                }
            }
        }
    }
}

// ---------------------------------------------------------------------------
// fp32 -> bf16 bulk conversion (weights), 4 elements/thread
// ---------------------------------------------------------------------------
__global__ __launch_bounds__(256) void cvt_bf16_kernel(
    const float* __restrict__ in, unsigned short* __restrict__ out)
{
    const long i4 = ((long)blockIdx.x * 256 + threadIdx.x) * 4;
    const float4 v = *(const float4*)(in + i4);
    unsigned int* o = (unsigned int*)(out + i4);
    o[0] = pack2_bf16(v.x, v.y);
    o[1] = pack2_bf16(v.z, v.w);
}

// ---------------------------------------------------------------------------
// RMSNorm (optional sinusoidal positional add); writes fp32 + bf16 copies.
// ---------------------------------------------------------------------------
__global__ __launch_bounds__(256) void norm_kernel(
    const float* __restrict__ in, const float* __restrict__ gain,
    const float* __restrict__ bias, float* __restrict__ outf,
    unsigned short* __restrict__ outb, int S, int Dd, int addPos)
{
    __shared__ float sm[32];
    const long row = blockIdx.x;
    const int  s   = (int)(row % S);
    const float* xr = in + row * (long)Dd;

    float pos = 0.0f;
    if (addPos) {
        const float f = (float)s;
        const float period = 1.0f / (powf(1000.0f, 2.0f * f / (float)S) + EPSF);
        pos = ((s & 1) == 0) ? sinf(period * f) : cosf(period * f);
    }

    float vals[8];
    float ss = 0.0f;
#pragma unroll
    for (int i = 0; i < 8; ++i) {   // Dd == 2048
        const int d = threadIdx.x + i * 256;
        const float v = xr[d] + pos;
        vals[i] = v;
        ss += v * v;
    }
    ss = blockReduceSum(ss, sm);
    const float inv = 1.0f / (sqrtf(ss / (float)Dd) + EPSF);

    float* orow = outf + row * (long)Dd;
    unsigned short* brow = outb + row * (long)Dd;
#pragma unroll
    for (int i = 0; i < 8; ++i) {
        const int d = threadIdx.x + i * 256;
        const float r = vals[i] * inv * gain[d] + bias[d];
        orow[d] = r;
        brow[d] = bf16_rne(r);
    }
}

// ---------------------------------------------------------------------------
// Row softmax: fp32 scores in, bf16 attention out.
// ---------------------------------------------------------------------------
__global__ __launch_bounds__(256) void softmax_kernel(
    const float* __restrict__ sc, unsigned short* __restrict__ attn, int S)
{
    __shared__ float sm[32];
    const float* r = sc + (long)blockIdx.x * (long)S;
    unsigned short* o = attn + (long)blockIdx.x * (long)S;
    float vals[8];
    float mx = -3.402823466e38f;
#pragma unroll
    for (int i = 0; i < 8; ++i) {   // S == 2048
        const float v = r[threadIdx.x + i * 256];
        vals[i] = v;
        mx = fmaxf(mx, v);
    }
    mx = blockReduceMax(mx, sm);
    float sum = 0.0f;
#pragma unroll
    for (int i = 0; i < 8; ++i) {
        const float e = __expf(vals[i] - mx);
        vals[i] = e;
        sum += e;
    }
    sum = blockReduceSum(sum, sm);
    const float inv = 1.0f / sum;
#pragma unroll
    for (int i = 0; i < 8; ++i)
        o[threadIdx.x + i * 256] = bf16_rne(vals[i] * inv);
}

// ---------------------------------------------------------------------------
// Final: h = rmsnorm(h_pre over H) ; a,g = split(h) ; out = a*sigmoid(g) + x2
// ---------------------------------------------------------------------------
__global__ __launch_bounds__(256) void glu_kernel(
    const float* __restrict__ hbuf, const float* __restrict__ gf,
    const float* __restrict__ bfn, const float* __restrict__ x2,
    float* __restrict__ out, int Dd, int Hh)
{
    __shared__ float sm[32];
    const long row = blockIdx.x;
    const float* hr = hbuf + row * (long)Hh;

    float ss = 0.0f;
#pragma unroll
    for (int i = 0; i < 16; ++i) {   // Hh == 4096
        const float v = hr[threadIdx.x + i * 256];
        ss += v * v;
    }
    ss = blockReduceSum(ss, sm);
    const float inv = 1.0f / (sqrtf(ss / (float)Hh) + EPSF);

    const float* x2r = x2 + row * (long)Dd;
    float* orow = out + row * (long)Dd;
#pragma unroll
    for (int i = 0; i < 8; ++i) {    // Dd == 2048
        const int d = threadIdx.x + i * 256;
        const float a = hr[d] * inv * gf[d] + bfn[d];
        const float g = hr[Dd + d] * inv * gf[Dd + d] + bfn[Dd + d];
        orow[d] = a * (1.0f / (1.0f + __expf(-g))) + x2r[d];
    }
}

// ---------------------------------------------------------------------------
// Host-side orchestration
// ---------------------------------------------------------------------------
template <int OM>
static inline void gemmL(hipStream_t st, const unsigned short* A, long lda,
                         const unsigned short* B, long ldb, void* out, long ldo,
                         int M, int N, int K, float alpha,
                         const float* bias, const float* res, long ldr)
{
    dim3 g((unsigned)(N / 128), (unsigned)(M / 128)), blk(256);
    gemm_tdm_wmma<OM><<<g, blk, 0, st>>>(A, lda, B, ldb, out, ldo, K, alpha,
                                         bias, res, ldr);
}

extern "C" void kernel_launch(void* const* d_in, const int* in_sizes, int n_in,
                              void* d_out, int out_size, void* d_ws, size_t ws_size,
                              hipStream_t stream)
{
    constexpr int Bn = 4, S = 2048, D = 2048, DK = 2048, H = 4096;
    constexpr int M = Bn * S;  // 8192

    const float* x   = (const float*)d_in[0];
    const float* Wq  = (const float*)d_in[1];
    const float* Wk  = (const float*)d_in[2];
    const float* Wv  = (const float*)d_in[3];
    const float* Wp  = (const float*)d_in[4];
    const float* bp  = (const float*)d_in[5];
    const float* g1  = (const float*)d_in[6];
    const float* b1  = (const float*)d_in[7];
    const float* g2  = (const float*)d_in[8];
    const float* b2  = (const float*)d_in[9];
    const float* Wf  = (const float*)d_in[10];
    const float* bf  = (const float*)d_in[11];
    const float* gf  = (const float*)d_in[12];
    const float* bfn = (const float*)d_in[13];

    char* ws = (char*)d_ws;
    const size_t F  = sizeof(float);
    const size_t MD = (size_t)M * D;

    float*          xn_f  = (float*)(ws);                         //  64 MiB (xn / x2)
    unsigned short* xn_b  = (unsigned short*)(ws + MD * F);       //  32 MiB (bf16)
    char*           blk0  = ws + MD * F + MD * 2;                 // 128 MiB block:
    unsigned short* qb    = (unsigned short*)(blk0);              //   q (later o)
    unsigned short* kb    = (unsigned short*)(blk0 + MD * 2);     //   k
    unsigned short* vtb   = (unsigned short*)(blk0 + MD * 4);     //   v^T
    unsigned short* attnb = (unsigned short*)(blk0 + MD * 6);     //   attn
    float*          h_f   = (float*)(blk0);                       //   h_pre overlays all 4
    float*          sc_f  = (float*)(blk0 + MD * 8);              //  64 MiB scores / attn-out
    unsigned short* wqb   = (unsigned short*)(blk0 + MD * 8 + MD * F);
    unsigned short* wkb   = wqb + (size_t)DK * D;
    unsigned short* wvb   = wkb + (size_t)DK * D;
    unsigned short* wpb   = wvb + (size_t)DK * D;
    unsigned short* wfb   = wpb + (size_t)D * DK;

    // 0) convert weights fp32 -> bf16 (once per call)
    cvt_bf16_kernel<<<dim3((DK * D) / 1024), dim3(256), 0, stream>>>(Wq, wqb);
    cvt_bf16_kernel<<<dim3((DK * D) / 1024), dim3(256), 0, stream>>>(Wk, wkb);
    cvt_bf16_kernel<<<dim3((DK * D) / 1024), dim3(256), 0, stream>>>(Wv, wvb);
    cvt_bf16_kernel<<<dim3((D * DK) / 1024), dim3(256), 0, stream>>>(Wp, wpb);
    cvt_bf16_kernel<<<dim3((H * D) / 1024),  dim3(256), 0, stream>>>(Wf, wfb);

    // 1) positional add + pre-norm -> xn (fp32 + bf16)
    norm_kernel<<<dim3(M), dim3(256), 0, stream>>>(x, g1, b1, xn_f, xn_b, S, D, 1);

    // 2) Q = xn*Wq^T, K = xn*Wk^T (bf16 out)
    gemmL<OM_BF16>(stream, xn_b, D, wqb, D, qb, DK, M, DK, D, 1.0f, nullptr, nullptr, 0);
    gemmL<OM_BF16>(stream, xn_b, D, wkb, D, kb, DK, M, DK, D, 1.0f, nullptr, nullptr, 0);

    // 3) V = xn*Wv^T stored transposed per batch: vt[b][n][s] (bf16)
    for (int b = 0; b < Bn; ++b)
        gemmL<OM_BF16_T>(stream, xn_b + (size_t)b * S * D, D, wvb, D,
                         vtb + (size_t)b * DK * S, S, S, DK, D, 1.0f, nullptr, nullptr, 0);

    // 4) scores[b] = (q[b]*k[b]^T)/sqrt(DK)  (fp32 out for softmax)
    const float scal = 1.0f / sqrtf((float)DK);
    for (int b = 0; b < Bn; ++b)
        gemmL<OM_F32>(stream, qb + (size_t)b * S * DK, DK, kb + (size_t)b * S * DK, DK,
                      sc_f + (size_t)b * S * S, S, S, S, DK, scal, nullptr, nullptr, 0);

    // 5) softmax rows -> bf16 attn
    softmax_kernel<<<dim3(M), dim3(256), 0, stream>>>(sc_f, attnb, S);

    // 6) o[b] = attn[b]*v[b] (NT vs vt) -> qb region (q is dead), bf16
    for (int b = 0; b < Bn; ++b)
        gemmL<OM_BF16>(stream, attnb + (size_t)b * S * S, S, vtb + (size_t)b * DK * S, S,
                       qb + (size_t)b * S * DK, DK, S, DK, S, 1.0f, nullptr, nullptr, 0);

    // 7) attn-out = o*Wp^T + bp + xn -> sc_f region (scores are dead), fp32
    gemmL<OM_F32_BIAS_RES>(stream, qb, DK, wpb, DK, sc_f, D, M, D, DK, 1.0f, bp, xn_f, D);

    // 8) x2 = rmsnorm(attn-out) -> xn regions (xn is dead)
    norm_kernel<<<dim3(M), dim3(256), 0, stream>>>(sc_f, g2, b2, xn_f, xn_b, S, D, 0);

    // 9) h_pre = x2*Wf^T + bf -> h_f (overlays q/k/vt/attn block, all dead)
    gemmL<OM_F32_BIAS>(stream, xn_b, D, wfb, D, h_f, H, M, H, D, 1.0f, bf, nullptr, 0);

    // 10) out = GLU(rmsnorm(h_pre)) + x2
    glu_kernel<<<dim3(M), dim3(256), 0, stream>>>(h_f, gf, bfn, xn_f,
                                                  (float*)d_out, D, H);
}
